// EF_encoder_18107582120620
// MI455X (gfx1250) — matrix operand order repaired and checked by reading, not compiled, not run on hardware
//
#include <hip/hip_runtime.h>

// ---------------------------------------------------------------------------
// CDNA5 / gfx1250 implementation of the EF point-cloud encoder-decoder.
// All dense linear layers (lin1d / lin2d / fc) run through a batched WMMA
// GEMM using v_wmma_f32_16x16x32_bf16 (bf16 A/B, f32 accumulate), with
// LDS-staged tiles (ds_load_b128 fragment reads, branch-free global loads).
// ---------------------------------------------------------------------------

typedef __attribute__((ext_vector_type(16))) __bf16        bf16x16;
typedef __attribute__((ext_vector_type(8)))  float         floatx8;
typedef __attribute__((ext_vector_type(8)))  unsigned int  uint32x8;

#define BQ 16   // batch
#define GQ 24   // dense-block growth channels
#define KQ 16   // kNN neighbors
#define MAXC 48 // max feature channels fed to kNN

#define TM 32   // block tile M
#define TN 64   // block tile N
#define TK 64   // block tile K (2 WMMA k-steps)
#define TKD (TK / 2)   // dwords along K (2 bf16 per dword)

__device__ __forceinline__ unsigned int pack_bf16(float a, float b)
{
    __bf16 x = (__bf16)a, y = (__bf16)b;
    unsigned short ux = __builtin_bit_cast(unsigned short, x);
    unsigned short uy = __builtin_bit_cast(unsigned short, y);
    return (unsigned int)ux | ((unsigned int)uy << 16);
}

// ---------------------------------------------------------------------------
// Generic batched GEMM:  Y[b, co+m, n] = act( sum_k W[m,k] * X[b, k, n] + bias[m] )
//   W: O x C row-major (shared across batch)
//   X: [nb, xC, Ncols], rows [0, C) consumed
//   Y: [nb, dstC, Ncols], written at channel offset co
// ---------------------------------------------------------------------------
__global__ void __launch_bounds__(256)
wmma_gemm_kernel(const float* __restrict__ W, const float* __restrict__ bias,
                 const float* __restrict__ X, float* __restrict__ Y,
                 int O, int C, int Ncols, int relu, int xC, int dstC, int co)
{
    __shared__ unsigned int As[TM * TKD];   // [m][kd]  packed bf16 K-pairs
    __shared__ unsigned int Bs[TN * TKD];   // [n][kd]  (transposed) packed pairs

    const int b    = blockIdx.z;
    const int tid  = threadIdx.x;
    const int lane = tid & 31;
    const int wid  = tid >> 5;
    const int g    = lane >> 4;      // half-wave group (0/1)
    const int l16  = lane & 15;

    const int mBlk = blockIdx.y * TM;
    const int nBlk = blockIdx.x * TN;

    const float* __restrict__ Xb = X + (size_t)b * xC * Ncols;

    floatx8 acc = {0.f, 0.f, 0.f, 0.f, 0.f, 0.f, 0.f, 0.f};

    const int mLoc = (wid & 1) * 16 + l16;   // this wave's A row in tile
    const int nLoc = (wid >> 1) * 16 + l16;  // this wave's B col in tile

    for (int k0 = 0; k0 < C; k0 += TK) {
        // ---- stage A tile (TM x TK), branch-free OOB -> 0 ----
#pragma unroll
        for (int it = tid; it < TM * TKD; it += 256) {
            const int m   = it / TKD;
            const int kd  = it % TKD;
            const int row = mBlk + m;
            const int kk  = k0 + 2 * kd;
            const bool ok0 = (row < O) && (kk     < C);
            const bool ok1 = (row < O) && (kk + 1 < C);
            float f0 = W[ok0 ? ((size_t)row * C + kk)     : 0]; if (!ok0) f0 = 0.f;
            float f1 = W[ok1 ? ((size_t)row * C + kk + 1) : 0]; if (!ok1) f1 = 0.f;
            As[it] = pack_bf16(f0, f1);
        }
        // ---- stage B tile (TK x TN), transposed into [n][kd] ----
#pragma unroll
        for (int it = tid; it < TN * TKD; it += 256) {
            const int kd  = it / TN;     // 0..TKD-1
            const int n   = it % TN;     // coalesced along n for each kd row
            const int col = nBlk + n;
            const int kk  = k0 + 2 * kd;
            const bool ok0 = (kk     < C) && (col < Ncols);
            const bool ok1 = (kk + 1 < C) && (col < Ncols);
            float f0 = Xb[ok0 ? ((size_t)kk       * Ncols + col) : 0]; if (!ok0) f0 = 0.f;
            float f1 = Xb[ok1 ? ((size_t)(kk + 1) * Ncols + col) : 0]; if (!ok1) f1 = 0.f;
            Bs[n * TKD + kd] = pack_bf16(f0, f1);
        }
        if (k0 + TK < C)
            __builtin_prefetch(&Xb[(size_t)(k0 + TK) * Ncols + nBlk], 0, 1);
        __syncthreads();

        // ---- two WMMA k-steps from LDS (ds_load_b128 fragments) ----
#pragma unroll
        for (int s = 0; s < 2; ++s) {
            const unsigned int* ap = &As[mLoc * TKD + s * 16];
            const uint4 a_lo = *(const uint4*)(ap + g * 4);       // K-pairs {0..3}+g*4
            const uint4 a_hi = *(const uint4*)(ap + 8 + g * 4);   // K-pairs {8..11}+g*4
            const unsigned int* bp = &Bs[nLoc * TKD + s * 16];
            const uint4 b_lo = *(const uint4*)(bp + g * 4);
            const uint4 b_hi = *(const uint4*)(bp + 8 + g * 4);

            uint32x8 aw = {a_lo.x, a_lo.y, a_lo.z, a_lo.w, a_hi.x, a_hi.y, a_hi.z, a_hi.w};
            uint32x8 bw = {b_lo.x, b_lo.y, b_lo.z, b_lo.w, b_hi.x, b_hi.y, b_hi.z, b_hi.w};
            bf16x16 af = __builtin_bit_cast(bf16x16, aw);
            bf16x16 bf = __builtin_bit_cast(bf16x16, bw);

            acc = __builtin_amdgcn_wmma_f32_16x16x32_bf16(false, af, false, bf,
                                                          (short)0, acc, false, false);
        }
        __syncthreads();
    }

    // ---- epilogue: bias + optional relu, predicated store ----
    const int m0 = mBlk + (wid & 1) * 16;
    const int n0 = nBlk + (wid >> 1) * 16;
    float* __restrict__ Yb = Y + ((size_t)b * dstC + co) * Ncols;
#pragma unroll
    for (int r = 0; r < 8; ++r) {
        const int m = m0 + r + (g << 3);     // C/D layout: VGPR r -> M = r (+8 for hi half)
        const int n = n0 + l16;
        if (m < O && n < Ncols) {
            float v = acc[r] + bias[m];
            if (relu) v = fmaxf(v, 0.f);
            Yb[(size_t)m * Ncols + n] = v;
        }
    }
}

// ---------------------------------------------------------------------------
// kNN in feature space (self-kNN over the N points of each batch).
// idxOut[b, n, 0..Kn) = indices of Kn nearest (incl. self), ascending distance.
// ---------------------------------------------------------------------------
__global__ void knn_feat_kernel(const float* __restrict__ X, int C, int N, int Kn,
                                int* __restrict__ idxOut)
{
    const int b = blockIdx.y;
    const int n = blockIdx.x * blockDim.x + threadIdx.x;
    if (n >= N) return;
    const float* __restrict__ Xb = X + (size_t)b * C * N;

    float f[MAXC];
    for (int c = 0; c < C; ++c) f[c] = Xb[(size_t)c * N + n];

    float kd[KQ]; int ki[KQ];
    for (int j = 0; j < Kn; ++j) { kd[j] = 3.4e38f; ki[j] = 0; }

    for (int m = 0; m < N; ++m) {
        float d = 0.f;
        for (int c = 0; c < C; ++c) {
            float t = f[c] - Xb[(size_t)c * N + m];
            d += t * t;
        }
        if (d < kd[Kn - 1]) {                 // stable insert (ties keep lower index)
            int j = Kn - 1;
            while (j > 0 && kd[j - 1] > d) { kd[j] = kd[j - 1]; ki[j] = ki[j - 1]; --j; }
            kd[j] = d; ki[j] = m;
        }
    }
    int* io = idxOut + ((size_t)b * N + n) * Kn;
    for (int j = 0; j < Kn; ++j) io[j] = ki[j];
}

// ---------------------------------------------------------------------------
// kNN in xyz space: queries Q[b,S,3] vs database P[b,M,3].
// ---------------------------------------------------------------------------
__global__ void knn_pts_kernel(const float* __restrict__ Q, const float* __restrict__ P,
                               int S, int M, int Kn,
                               int* __restrict__ idxOut, float* __restrict__ distOut)
{
    const int b = blockIdx.y;
    const int s = blockIdx.x * blockDim.x + threadIdx.x;
    if (s >= S) return;
    const float* q = Q + ((size_t)b * S + s) * 3;
    const float qx = q[0], qy = q[1], qz = q[2];
    const float* __restrict__ Pb = P + (size_t)b * M * 3;

    float kd[KQ]; int ki[KQ];
    for (int j = 0; j < Kn; ++j) { kd[j] = 3.4e38f; ki[j] = 0; }

    for (int m = 0; m < M; ++m) {
        float dx = qx - Pb[(size_t)m * 3 + 0];
        float dy = qy - Pb[(size_t)m * 3 + 1];
        float dz = qz - Pb[(size_t)m * 3 + 2];
        float d = dx * dx + dy * dy + dz * dz;
        if (d < kd[Kn - 1]) {
            int j = Kn - 1;
            while (j > 0 && kd[j - 1] > d) { kd[j] = kd[j - 1]; ki[j] = ki[j - 1]; --j; }
            kd[j] = d; ki[j] = m;
        }
    }
    int* io = idxOut + ((size_t)b * S + s) * Kn;
    for (int j = 0; j < Kn; ++j) io[j] = ki[j];
    if (distOut) {
        float* dn = distOut + ((size_t)b * S + s) * Kn;
        for (int j = 0; j < Kn; ++j) dn[j] = kd[j];
    }
}

// ---------------------------------------------------------------------------
// Farthest point sampling: one block per batch, dist table in LDS, serial
// argmax chain with a block tree-reduction per step.
// ---------------------------------------------------------------------------
__global__ void __launch_bounds__(256)
fps_kernel(const float* __restrict__ P, int M, int S, int* __restrict__ outIdx)
{
    __shared__ float dists[2048];
    __shared__ float rv[256];
    __shared__ int   ri[256];
    __shared__ int   s_last;

    const int b = blockIdx.x;
    const float* __restrict__ Pb = P + (size_t)b * M * 3;
    const int t = threadIdx.x;

    for (int m = t; m < M; m += 256) dists[m] = 1e10f;
    if (t == 0) { s_last = 0; outIdx[(size_t)b * S] = 0; }
    __syncthreads();

    for (int i = 1; i < S; ++i) {
        const int last = s_last;
        const float lx = Pb[(size_t)last * 3 + 0];
        const float ly = Pb[(size_t)last * 3 + 1];
        const float lz = Pb[(size_t)last * 3 + 2];
        float bestv = -1.f; int besti = 0x7fffffff;
        for (int m = t; m < M; m += 256) {
            float dx = Pb[(size_t)m * 3 + 0] - lx;
            float dy = Pb[(size_t)m * 3 + 1] - ly;
            float dz = Pb[(size_t)m * 3 + 2] - lz;
            float d = dx * dx + dy * dy + dz * dz;
            float dm = fminf(dists[m], d);
            dists[m] = dm;
            if (dm > bestv) { bestv = dm; besti = m; }
        }
        rv[t] = bestv; ri[t] = besti;
        __syncthreads();
        for (int off = 128; off > 0; off >>= 1) {
            if (t < off) {
                if (rv[t + off] > rv[t] ||
                    (rv[t + off] == rv[t] && ri[t + off] < ri[t])) {
                    rv[t] = rv[t + off]; ri[t] = ri[t + off];
                }
            }
            __syncthreads();
        }
        if (t == 0) { s_last = ri[0]; outIdx[(size_t)b * S + i] = ri[0]; }
        __syncthreads();
    }
}

// --------------------------- elementwise helpers ---------------------------

__global__ void pc_transpose_kernel(const float* __restrict__ x, float* __restrict__ P, int N)
{
    size_t tid = (size_t)blockIdx.x * blockDim.x + threadIdx.x;
    size_t total = (size_t)BQ * N * 3;
    if (tid >= total) return;
    int c = (int)(tid % 3); size_t t = tid / 3;
    int n = (int)(t % N);   int b = (int)(t / N);
    P[((size_t)b * N + n) * 3 + c] = x[((size_t)b * 3 + c) * N + n];
}

__global__ void gather_pts_kernel(const float* __restrict__ P, const int* __restrict__ pidx,
                                  int S, int M, float* __restrict__ Q)
{
    size_t tid = (size_t)blockIdx.x * blockDim.x + threadIdx.x;
    size_t total = (size_t)BQ * S * 3;
    if (tid >= total) return;
    int d = (int)(tid % 3); size_t t = tid / 3;
    int s = (int)(t % S);   int b = (int)(t / S);
    int j = pidx[(size_t)b * S + s];
    Q[((size_t)b * S + s) * 3 + d] = P[((size_t)b * M + j) * 3 + d];
}

// E[b, c, n, k]: c<C -> X[b,c,idx[n,k]] - X[b,c,n] ; c>=C -> X[b,c-C,n]
__global__ void edge_feat_kernel(const float* __restrict__ X, const int* __restrict__ idx,
                                 int C, int N, int Kn, float* __restrict__ E)
{
    size_t tid = (size_t)blockIdx.x * blockDim.x + threadIdx.x;
    size_t total = (size_t)BQ * 2 * C * N * Kn;
    if (tid >= total) return;
    int k = (int)(tid % Kn); size_t t = tid / Kn;
    int n = (int)(t % N); t /= N;
    int c = (int)(t % (2 * C));
    int b = (int)(t / (2 * C));
    const float* __restrict__ Xb = X + (size_t)b * C * N;
    float v;
    if (c < C) {
        int j = idx[((size_t)b * N + n) * Kn + k];
        v = Xb[(size_t)c * N + j] - Xb[(size_t)c * N + n];
    } else {
        v = Xb[(size_t)(c - C) * N + n];
    }
    E[((size_t)b * 2 * C + c) * ((size_t)N * Kn) + (size_t)n * Kn + k] = v;
}

// Y[b, co+c, n, k] = X[b, c, n]
__global__ void bcast_nk_kernel(const float* __restrict__ X, int C, int N, int Kn,
                                float* __restrict__ Y, int dstC, int co)
{
    size_t tid = (size_t)blockIdx.x * blockDim.x + threadIdx.x;
    size_t total = (size_t)BQ * C * N * Kn;
    if (tid >= total) return;
    int k = (int)(tid % Kn); size_t t = tid / Kn;
    int n = (int)(t % N); t /= N;
    int c = (int)(t % C); int b = (int)(t / C);
    Y[((size_t)b * dstC + co + c) * ((size_t)N * Kn) + (size_t)n * Kn + k] =
        X[((size_t)b * C + c) * N + n];
}

// dst[b, co+c, n] = (relu?) max_k Y[b, c, n, k]
__global__ void maxk_kernel(const float* __restrict__ Y, int Cc, int N, int Kn,
                            float* __restrict__ dst, int dstC, int co, int relu)
{
    size_t tid = (size_t)blockIdx.x * blockDim.x + threadIdx.x;
    size_t total = (size_t)BQ * Cc * N;
    if (tid >= total) return;
    int n = (int)(tid % N); size_t t = tid / N;
    int c = (int)(t % Cc);  int b = (int)(t / Cc);
    const float* p = Y + ((size_t)b * Cc + c) * ((size_t)N * Kn) + (size_t)n * Kn;
    float m = p[0];
    for (int k = 1; k < Kn; ++k) m = fmaxf(m, p[k]);
    if (relu) m = fmaxf(m, 0.f);
    dst[((size_t)b * dstC + co + c) * N + n] = m;
}

// dst[b, co+c, n] = src[b, c, n]
__global__ void copy_ch_kernel(float* __restrict__ dst, int dstC, int co,
                               const float* __restrict__ src, int srcC, int Cc, int Ncols)
{
    size_t tid = (size_t)blockIdx.x * blockDim.x + threadIdx.x;
    size_t total = (size_t)BQ * Cc * Ncols;
    if (tid >= total) return;
    int n = (int)(tid % Ncols); size_t t = tid / Ncols;
    int c = (int)(t % Cc);      int b = (int)(t / Cc);
    dst[((size_t)b * dstC + co + c) * Ncols + n] = src[((size_t)b * srcC + c) * Ncols + n];
}

// dst[b, co+c, s] = F[b, c, pidx[b,s]]
__global__ void center_gather_kernel(const float* __restrict__ F, int Cs, int N,
                                     const int* __restrict__ pidx, int S,
                                     float* __restrict__ dst, int dstC, int co)
{
    size_t tid = (size_t)blockIdx.x * blockDim.x + threadIdx.x;
    size_t total = (size_t)BQ * Cs * S;
    if (tid >= total) return;
    int s = (int)(tid % S); size_t t = tid / S;
    int c = (int)(t % Cs);  int b = (int)(t / Cs);
    int j = pidx[(size_t)b * S + s];
    dst[((size_t)b * dstC + co + c) * S + s] = F[((size_t)b * Cs + c) * N + j];
}

// dst[b, co+c, s] = max_k F[b, c, pnidx[b,s,k]]
__global__ void nbmax_gather_kernel(const float* __restrict__ F, int Cs, int N,
                                    const int* __restrict__ pnidx, int S, int Kn,
                                    float* __restrict__ dst, int dstC, int co)
{
    size_t tid = (size_t)blockIdx.x * blockDim.x + threadIdx.x;
    size_t total = (size_t)BQ * Cs * S;
    if (tid >= total) return;
    int s = (int)(tid % S); size_t t = tid / S;
    int c = (int)(t % Cs);  int b = (int)(t / Cs);
    const int* id = pnidx + ((size_t)b * S + s) * Kn;
    const float* Fr = F + ((size_t)b * Cs + c) * N;
    float m = Fr[id[0]];
    for (int k = 1; k < Kn; ++k) m = fmaxf(m, Fr[id[k]]);
    dst[((size_t)b * dstC + co + c) * S + s] = m;
}

// gT[o, b] = max_n Z[b, o, n]
__global__ void rowmax_T_kernel(const float* __restrict__ Z, int O, int Ncols,
                                float* __restrict__ gT)
{
    size_t tid = (size_t)blockIdx.x * blockDim.x + threadIdx.x;
    size_t total = (size_t)BQ * O;
    if (tid >= total) return;
    int o = (int)(tid % O); int b = (int)(tid / O);
    const float* p = Z + ((size_t)b * O + o) * Ncols;
    float m = p[0];
    for (int n = 1; n < Ncols; ++n) m = fmaxf(m, p[n]);
    gT[(size_t)o * BQ + b] = m;
}

// dst[b, co+o, n] = g[o, b]   (broadcast global feature across points)
__global__ void bcast_vec_kernel(const float* __restrict__ g, int O,
                                 float* __restrict__ dst, int dstC, int co, int Ncols)
{
    size_t tid = (size_t)blockIdx.x * blockDim.x + threadIdx.x;
    size_t total = (size_t)BQ * O * Ncols;
    if (tid >= total) return;
    int n = (int)(tid % Ncols); size_t t = tid / Ncols;
    int o = (int)(t % O);       int b = (int)(t / O);
    dst[((size_t)b * dstC + co + o) * Ncols + n] = g[(size_t)o * BQ + b];
}

// 3-NN inverse-distance interpolation:
// dst[b, co+c, s] = sum_j w_j F[b, c, idx3[b,s,j]],  w_j = (1/max(d2,1e-10)) / sum
__global__ void interp_kernel(const float* __restrict__ F, int Cs, int Ssrc,
                              const int* __restrict__ idx3, const float* __restrict__ d3,
                              int Sdst, float* __restrict__ dst, int dstC, int co)
{
    size_t tid = (size_t)blockIdx.x * blockDim.x + threadIdx.x;
    size_t total = (size_t)BQ * Cs * Sdst;
    if (tid >= total) return;
    int s = (int)(tid % Sdst); size_t t = tid / Sdst;
    int c = (int)(t % Cs);     int b = (int)(t / Cs);
    const int*   id = idx3 + ((size_t)b * Sdst + s) * 3;
    const float* dd = d3   + ((size_t)b * Sdst + s) * 3;
    float w0 = 1.f / fmaxf(dd[0], 1e-10f);
    float w1 = 1.f / fmaxf(dd[1], 1e-10f);
    float w2 = 1.f / fmaxf(dd[2], 1e-10f);
    float wsum = w0 + w1 + w2;
    const float* Fr = F + ((size_t)b * Cs + c) * Ssrc;
    float v = (w0 * Fr[id[0]] + w1 * Fr[id[1]] + w2 * Fr[id[2]]) / wsum;
    dst[((size_t)b * dstC + co + c) * Sdst + s] = v;
}

// ---------------------------------------------------------------------------
// Host orchestration
// ---------------------------------------------------------------------------
struct LinP { const float* w; const float* b; };

extern "C" void kernel_launch(void* const* d_in, const int* in_sizes, int n_in,
                              void* d_out, int out_size, void* d_ws, size_t ws_size,
                              hipStream_t stream)
{
    (void)n_in; (void)out_size; (void)ws_size;

    // ---- map inputs (handles both insertion-order and sorted-key flatten) ----
    const float* x = nullptr;
    LinP conv1{}, conv2{}, conv3{}, conv4{}, conv5{}, conv6{}, conv7{}, conv8{};
    LinP dc1f{}, dc1s1{}, dc1s2{}, dc2f{}, dc2s1{}, dc2s2{};
    LinP dc3f{}, dc3s1{}, dc3s2{}, dc4f{}, dc4s1{}, dc4s2{};
    LinP gfp{}, fc1{}, fc2{};

    auto F = [&](int i) { return (const float*)d_in[i]; };
    if (in_sizes[0] == BQ * 3 * 2048) {
        // insertion order: x first, then params {w,b}
        int i = 0;
        x = F(i++);
        auto L = [&](LinP& l) { l.w = F(i++); l.b = F(i++); };
        L(conv1);
        L(dc1f); L(dc1s1); L(dc1s2);
        L(conv2);
        L(dc2f); L(dc2s1); L(dc2s2);
        L(conv3);
        L(dc3f); L(dc3s1); L(dc3s2);
        L(conv4);
        L(dc4f); L(dc4s1); L(dc4s2);
        L(gfp); L(fc1); L(fc2);
        L(conv5); L(conv6); L(conv7); L(conv8);
    } else {
        // JAX sorted-key order: params first ({b,w} per lin), x last
        int i = 0;
        auto L = [&](LinP& l) { l.b = F(i++); l.w = F(i++); };
        L(conv1); L(conv2); L(conv3); L(conv4);
        L(conv5); L(conv6); L(conv7); L(conv8);
        L(dc1f); L(dc1s1); L(dc1s2);
        L(dc2f); L(dc2s1); L(dc2s2);
        L(dc3f); L(dc3s1); L(dc3s2);
        L(dc4f); L(dc4s1); L(dc4s2);
        L(fc1); L(fc2); L(gfp);
        x = F(i++);
    }

    // ---- workspace bump allocator ----
    float* wsf = (float*)d_ws;
    size_t off = 0;
    auto alloc = [&](size_t n) -> float* {
        float* p = wsf + off;
        off += (n + 63) & ~(size_t)63;
        return p;
    };

    float* P1    = alloc((size_t)BQ * 2048 * 3);
    float* P2    = alloc((size_t)BQ * 1024 * 3);
    float* P3    = alloc((size_t)BQ * 256 * 3);
    float* P4    = alloc((size_t)BQ * 64 * 3);
    float* X0    = alloc((size_t)BQ * 24 * 2048);
    float* X1    = alloc((size_t)BQ * 120 * 2048);
    float* X1D   = alloc((size_t)BQ * 240 * 1024);
    float* X2    = alloc((size_t)BQ * 48 * 1024);
    float* X2cat = alloc((size_t)BQ * 360 * 1024);
    float* X2D   = alloc((size_t)BQ * 720 * 256);
    float* X3    = alloc((size_t)BQ * 48 * 256);
    float* X3cat = alloc((size_t)BQ * 840 * 256);
    float* X3D   = alloc((size_t)BQ * 1680 * 64);
    float* X4    = alloc((size_t)BQ * 48 * 64);
    float* X4cat = alloc((size_t)BQ * 1800 * 64);
    float* Zgf   = alloc((size_t)BQ * 1024 * 64);
    float* gfT   = alloc((size_t)1024 * BQ);
    float* f1T   = alloc((size_t)512 * BQ);
    float* f2T   = alloc((size_t)1024 * BQ);
    float* X5in  = alloc((size_t)BQ * 2824 * 64);
    float* X5    = alloc((size_t)BQ * 1024 * 64);
    float* X6in  = alloc((size_t)BQ * 1864 * 256);
    float* X6    = alloc((size_t)BQ * 768 * 256);
    float* X7in  = alloc((size_t)BQ * 1128 * 1024);
    float* X7    = alloc((size_t)BQ * 512 * 1024);
    float* X8in  = alloc((size_t)BQ * 632 * 2048);
    int* iF1   = (int*)alloc((size_t)BQ * 2048 * KQ);
    int* iF2   = (int*)alloc((size_t)BQ * 1024 * KQ);
    int* iF3   = (int*)alloc((size_t)BQ * 256 * KQ);
    int* iF4   = (int*)alloc((size_t)BQ * 64 * KQ);
    int* pidx1 = (int*)alloc((size_t)BQ * 1024);
    int* pidx2 = (int*)alloc((size_t)BQ * 256);
    int* pidx3 = (int*)alloc((size_t)BQ * 64);
    int* pnn1  = (int*)alloc((size_t)BQ * 1024 * KQ);
    int* pnn2  = (int*)alloc((size_t)BQ * 256 * KQ);
    int* pnn3  = (int*)alloc((size_t)BQ * 64 * KQ);
    int*   i43 = (int*)alloc((size_t)BQ * 256 * 3);
    float* d43 = alloc((size_t)BQ * 256 * 3);
    int*   i32 = (int*)alloc((size_t)BQ * 1024 * 3);
    float* d32 = alloc((size_t)BQ * 1024 * 3);
    int*   i21 = (int*)alloc((size_t)BQ * 2048 * 3);
    float* d21 = alloc((size_t)BQ * 2048 * 3);
    float* arena = wsf + off;   // reused per dense-conv stage for E / Y tensors

    // ---- launch helpers ----
    auto blocks = [](size_t total) { return (unsigned)((total + 255) / 256); };

    auto gemm = [&](const LinP& p, const float* Xp, float* Yp, int O, int C, int Ncols,
                    int nb, int relu, int xC, int dstC, int co) {
        dim3 grid((Ncols + TN - 1) / TN, (O + TM - 1) / TM, nb);
        wmma_gemm_kernel<<<grid, 256, 0, stream>>>(p.w, p.b, Xp, Yp, O, C, Ncols,
                                                   relu, xC, dstC, co);
    };

    // dense_conv: kNN -> edge feat -> first/s1/s2 WMMA GEMMs -> max over k (+relu)
    auto run_dense = [&](const LinP& pf, const LinP& p1, const LinP& p2,
                         const float* Xin, int C, int Ncur,
                         float* dst, int dstC, int* idxbuf) {
        const int Cy = C + 3 * GQ;
        const int ncolsK = Ncur * KQ;
        float* E  = arena;
        float* Yb = arena + (((size_t)BQ * 2 * C * ncolsK + 63) & ~(size_t)63);

        knn_feat_kernel<<<dim3((Ncur + 127) / 128, BQ), 128, 0, stream>>>(
            Xin, C, Ncur, KQ, idxbuf);
        edge_feat_kernel<<<blocks((size_t)BQ * 2 * C * ncolsK), 256, 0, stream>>>(
            Xin, idxbuf, C, Ncur, KQ, E);
        gemm(pf, E, Yb, GQ, 2 * C, ncolsK, BQ, 1, 2 * C, Cy, 0);
        bcast_nk_kernel<<<blocks((size_t)BQ * C * ncolsK), 256, 0, stream>>>(
            Xin, C, Ncur, KQ, Yb, Cy, GQ);
        gemm(p1, Yb, Yb, GQ, GQ + C, ncolsK, BQ, 1, Cy, Cy, GQ + C);
        gemm(p2, Yb, Yb, GQ, C + 2 * GQ, ncolsK, BQ, 0, Cy, Cy, C + 2 * GQ);
        maxk_kernel<<<blocks((size_t)BQ * Cy * Ncur), 256, 0, stream>>>(
            Yb, Cy, Ncur, KQ, dst, dstC, 0, 1);
    };

    // ================= forward pass =================
    pc_transpose_kernel<<<blocks((size_t)BQ * 2048 * 3), 256, 0, stream>>>(x, P1, 2048);

    // x0 = relu(conv1(x))
    gemm(conv1, x, X0, 24, 3, 2048, BQ, 1, 3, 24, 0);

    // x1 = concat(relu(dense_conv(dc1, x0)), x0)  -> 120ch @ 2048
    run_dense(dc1f, dc1s1, dc1s2, X0, 24, 2048, X1, 120, iF1);
    copy_ch_kernel<<<blocks((size_t)BQ * 24 * 2048), 256, 0, stream>>>(
        X1, 120, 96, X0, 24, 24, 2048);

    // eps -> x1d (240ch @ 1024), pc2
    fps_kernel<<<BQ, 256, 0, stream>>>(P1, 2048, 1024, pidx1);
    gather_pts_kernel<<<blocks((size_t)BQ * 1024 * 3), 256, 0, stream>>>(P1, pidx1, 1024, 2048, P2);
    knn_pts_kernel<<<dim3((1024 + 127) / 128, BQ), 128, 0, stream>>>(
        P2, P1, 1024, 2048, KQ, pnn1, nullptr);
    center_gather_kernel<<<blocks((size_t)BQ * 120 * 1024), 256, 0, stream>>>(
        X1, 120, 2048, pidx1, 1024, X1D, 240, 0);
    nbmax_gather_kernel<<<blocks((size_t)BQ * 120 * 1024), 256, 0, stream>>>(
        X1, 120, 2048, pnn1, 1024, KQ, X1D, 240, 120);

    // x2 = relu(conv2(x1d)); dense; concat -> 360ch @ 1024
    gemm(conv2, X1D, X2, 48, 240, 1024, BQ, 1, 240, 48, 0);
    run_dense(dc2f, dc2s1, dc2s2, X2, 48, 1024, X2cat, 360, iF2);
    copy_ch_kernel<<<blocks((size_t)BQ * 240 * 1024), 256, 0, stream>>>(
        X2cat, 360, 120, X1D, 240, 240, 1024);

    // eps -> x2d (720ch @ 256), pc3
    fps_kernel<<<BQ, 256, 0, stream>>>(P2, 1024, 256, pidx2);
    gather_pts_kernel<<<blocks((size_t)BQ * 256 * 3), 256, 0, stream>>>(P2, pidx2, 256, 1024, P3);
    knn_pts_kernel<<<dim3((256 + 127) / 128, BQ), 128, 0, stream>>>(
        P3, P2, 256, 1024, KQ, pnn2, nullptr);
    center_gather_kernel<<<blocks((size_t)BQ * 360 * 256), 256, 0, stream>>>(
        X2cat, 360, 1024, pidx2, 256, X2D, 720, 0);
    nbmax_gather_kernel<<<blocks((size_t)BQ * 360 * 256), 256, 0, stream>>>(
        X2cat, 360, 1024, pnn2, 256, KQ, X2D, 720, 360);

    // x3 = relu(conv3(x2d)); dense; concat -> 840ch @ 256
    gemm(conv3, X2D, X3, 48, 720, 256, BQ, 1, 720, 48, 0);
    run_dense(dc3f, dc3s1, dc3s2, X3, 48, 256, X3cat, 840, iF3);
    copy_ch_kernel<<<blocks((size_t)BQ * 720 * 256), 256, 0, stream>>>(
        X3cat, 840, 120, X2D, 720, 720, 256);

    // eps -> x3d (1680ch @ 64), pc4
    fps_kernel<<<BQ, 256, 0, stream>>>(P3, 256, 64, pidx3);
    gather_pts_kernel<<<blocks((size_t)BQ * 64 * 3), 256, 0, stream>>>(P3, pidx3, 64, 256, P4);
    knn_pts_kernel<<<dim3(1, BQ), 128, 0, stream>>>(P4, P3, 64, 256, KQ, pnn3, nullptr);
    center_gather_kernel<<<blocks((size_t)BQ * 840 * 64), 256, 0, stream>>>(
        X3cat, 840, 256, pidx3, 64, X3D, 1680, 0);
    nbmax_gather_kernel<<<blocks((size_t)BQ * 840 * 64), 256, 0, stream>>>(
        X3cat, 840, 256, pnn3, 64, KQ, X3D, 1680, 840);

    // x4 = relu(conv4(x3d)); dense; concat -> 1800ch @ 64
    gemm(conv4, X3D, X4, 48, 1680, 64, BQ, 1, 1680, 48, 0);
    run_dense(dc4f, dc4s1, dc4s2, X4, 48, 64, X4cat, 1800, iF4);
    copy_ch_kernel<<<blocks((size_t)BQ * 1680 * 64), 256, 0, stream>>>(
        X4cat, 1800, 120, X3D, 1680, 1680, 64);

    // global feature: gf = max_n(lin(gf, x4)); relu(fc1); relu(fc2)
    gemm(gfp, X4cat, Zgf, 1024, 1800, 64, BQ, 0, 1800, 1024, 0);
    rowmax_T_kernel<<<blocks((size_t)BQ * 1024), 256, 0, stream>>>(Zgf, 1024, 64, gfT);
    gemm(fc1, gfT, f1T, 512, 1024, BQ, 1, 1, 1024, 512, 0);
    gemm(fc2, f1T, f2T, 1024, 512, BQ, 1, 1, 512, 1024, 0);

    // conv5 on concat(gf_broadcast, x4) -> 1024ch @ 64
    bcast_vec_kernel<<<blocks((size_t)BQ * 1024 * 64), 256, 0, stream>>>(
        f2T, 1024, X5in, 2824, 0, 64);
    copy_ch_kernel<<<blocks((size_t)BQ * 1800 * 64), 256, 0, stream>>>(
        X5in, 2824, 1024, X4cat, 1800, 1800, 64);
    gemm(conv5, X5in, X5, 1024, 2824, 64, BQ, 1, 2824, 1024, 0);

    // up: pc4(64) -> pc3(256); conv6 on concat(x3cat, up)
    knn_pts_kernel<<<dim3((256 + 127) / 128, BQ), 128, 0, stream>>>(P3, P4, 256, 64, 3, i43, d43);
    copy_ch_kernel<<<blocks((size_t)BQ * 840 * 256), 256, 0, stream>>>(
        X6in, 1864, 0, X3cat, 840, 840, 256);
    interp_kernel<<<blocks((size_t)BQ * 1024 * 256), 256, 0, stream>>>(
        X5, 1024, 64, i43, d43, 256, X6in, 1864, 840);
    gemm(conv6, X6in, X6, 768, 1864, 256, BQ, 1, 1864, 768, 0);

    // up: pc3(256) -> pc2(1024); conv7 on concat(x2cat, up)
    knn_pts_kernel<<<dim3((1024 + 127) / 128, BQ), 128, 0, stream>>>(P2, P3, 1024, 256, 3, i32, d32);
    copy_ch_kernel<<<blocks((size_t)BQ * 360 * 1024), 256, 0, stream>>>(
        X7in, 1128, 0, X2cat, 360, 360, 1024);
    interp_kernel<<<blocks((size_t)BQ * 768 * 1024), 256, 0, stream>>>(
        X6, 768, 256, i32, d32, 1024, X7in, 1128, 360);
    gemm(conv7, X7in, X7, 512, 1128, 1024, BQ, 1, 1128, 512, 0);

    // up: pc2(1024) -> pc1(2048); conv8 on concat(x1, up) -> output (no relu)
    knn_pts_kernel<<<dim3((2048 + 127) / 128, BQ), 128, 0, stream>>>(P1, P2, 2048, 1024, 3, i21, d21);
    copy_ch_kernel<<<blocks((size_t)BQ * 120 * 2048), 256, 0, stream>>>(
        X8in, 632, 0, X1, 120, 120, 2048);
    interp_kernel<<<blocks((size_t)BQ * 512 * 2048), 256, 0, stream>>>(
        X7, 512, 1024, i21, d21, 2048, X8in, 632, 120);
    gemm(conv8, X8in, (float*)d_out, 256, 632, 2048, BQ, 0, 632, 256, 0);
}
